// ClassifyModel_81209241633051
// MI455X (gfx1250) — matrix operand order repaired and checked by reading, not compiled, not run on hardware
//
#include <hip/hip_runtime.h>
#include <hip/hip_bf16.h>
#include <math.h>
#include <stdint.h>

// ---------------------------------------------------------------------------
// MI455X (gfx1250). All heavy GEMMs on the WMMA pipe: bf16 in / f32 acc
// (v_wmma_f32_16x16x32_bf16, wave32). Weights pre-transposed to [N,K].
// Tile staging via the Tensor Data Mover (TENSOR_LOAD_TO_LDS + TENSORcnt)
// when the builtin is available; register-staged double buffering otherwise.
// ---------------------------------------------------------------------------

typedef __bf16 bf16;
typedef __bf16 bf16x8 __attribute__((ext_vector_type(8)));
typedef __bf16 v16bf  __attribute__((ext_vector_type(16)));
typedef float  v8f    __attribute__((ext_vector_type(8)));

#if __has_builtin(__builtin_amdgcn_tensor_load_to_lds) && \
    __has_builtin(__builtin_amdgcn_s_wait_tensorcnt)
#define USE_TDM 1
#else
#define USE_TDM 0
#endif

#define BBATCH 8192
#define DD     768
#define DD2    1536
#define DD4    3072
#define NLAB   2

static constexpr size_t BD  = (size_t)BBATCH * DD;     // B*768
static constexpr size_t BD2 = (size_t)BBATCH * DD2;    // B*1536

// ---------------- GEMM: C[M,N] (f32) = A[M,K] (bf16) x Wt[N,K] (bf16) ------
#define BM 128
#define BN 128
#define BK 64
#define LDSS 72   // padded LDS row stride (elements): 144B rows, 16B aligned

#if USE_TDM
typedef unsigned int u32x4 __attribute__((ext_vector_type(4)));
typedef int          i32x8 __attribute__((ext_vector_type(8)));
typedef int          i32x4 __attribute__((ext_vector_type(4)));

// Issue one TDM load of a 128x64 bf16 tile (row stride = lda elements) into
// LDS at lds_byte_addr, padding each 128B row with 16B -> LDSS=72 elem rows.
__device__ __forceinline__ void tdm_load_tile(const bf16* gsrc, int lda,
                                              unsigned lds_byte_addr) {
    unsigned long long ga = (unsigned long long)(uintptr_t)gsrc;
    u32x4 g0 = { 1u,                                   // count=1 (valid D#)
                 lds_byte_addr,                        // lds_addr
                 (unsigned)(ga & 0xFFFFFFFFu),         // global_addr[31:0]
                 (unsigned)((ga >> 32) & 0x01FFFFFFu)  // global_addr[56:32]
                 | (2u << 30) };                       // type=2 ("image")
    i32x8 g1 = { (int)((1u << 16)      // data_size: 2 bytes
                 | (1u << 20)          // pad_enable
                 | (4u << 22)          // pad_interval: 32 dwords (=64 bf16 row)
                 | (3u << 25)),        // pad_amount: 4 dwords (=8 bf16 pad)
                 (int)(64u << 16),     // tensor_dim0 = 64 (abar addr = 0)
                 (int)(128u << 16),    // tensor_dim1 = 128
                 (int)(64u << 16),     // tile_dim0 = 64
                 128,                  // tile_dim1 = 128 (tile_dim2 = 0)
                 lda,                  // tensor_dim0_stride[31:0]
                 0, 0 };               // stride hi / dim1_stride (2D: ignored)
    i32x4 gz4 = { 0, 0, 0, 0 };
    i32x8 gz8 = { 0, 0, 0, 0, 0, 0, 0, 0 };
    __builtin_amdgcn_tensor_load_to_lds(g0, g1, gz4, gz4, gz8, 0);
}

__device__ __forceinline__ unsigned lds_addr32(const void* p) {
    // flat LDS address: low 32 bits hold the LDS byte offset
    return (unsigned)(uintptr_t)p;
}
#endif

__global__ __launch_bounds__(256)
void gemm_bf16_kernel(const bf16* __restrict__ A, int lda,
                      const bf16* __restrict__ Bt, int ldb,   // transposed [N,K]
                      float* __restrict__ C, int ldc,
                      int K, int accum)
{
    __shared__ __align__(16) bf16 As[2][BM * LDSS];
    __shared__ __align__(16) bf16 Bs[2][BN * LDSS];

    const int tid  = threadIdx.x;
    const int wave = tid >> 5;        // 0..7
    const int lane = tid & 31;
    const int half = lane >> 4;       // 0/1 (lane-half selects K-subrange)
    const int lrow = lane & 15;

    const int wm = (wave >> 2) * 64;  // wave M offset in block tile (0/64)
    const int wn = (wave & 3) * 32;   // wave N offset (0,32,64,96)

    const size_t blockM = (size_t)blockIdx.x * BM;
    const size_t blockN = (size_t)blockIdx.y * BN;

    v8f acc[4][2];
#pragma unroll
    for (int i = 0; i < 4; i++)
#pragma unroll
        for (int j = 0; j < 2; j++)
#pragma unroll
            for (int e = 0; e < 8; e++) acc[i][j][e] = 0.0f;

#if USE_TDM
    const bf16* gAt = A  + blockM * (size_t)lda;   // tile row base
    const bf16* gBt = Bt + blockN * (size_t)ldb;
    unsigned ldsA[2] = { lds_addr32(&As[0][0]), lds_addr32(&As[1][0]) };
    unsigned ldsB[2] = { lds_addr32(&Bs[0][0]), lds_addr32(&Bs[1][0]) };

    // prologue: DMA tile 0 into buf 0 (wave0: A, wave1: B; EXEC-independent)
    if (wave == 0)      tdm_load_tile(gAt, lda, ldsA[0]);
    else if (wave == 1) tdm_load_tile(gBt, ldb, ldsB[0]);
#else
    // staging: each thread owns one row (of 128) and half its 64 K-columns
    const int lr = tid >> 1;          // 0..127
    const int lc = (tid & 1) * 32;    // 0 or 32
    const bf16* gA = A  + (blockM + lr) * (size_t)lda + lc;
    const bf16* gB = Bt + (blockN + lr) * (size_t)ldb + lc;

    bf16x8 ra[4], rb[4];
#pragma unroll
    for (int q = 0; q < 4; q++) {
        ra[q] = *(const bf16x8*)(gA + 8 * q);
        rb[q] = *(const bf16x8*)(gB + 8 * q);
    }
#pragma unroll
    for (int q = 0; q < 4; q++) {
        *(bf16x8*)&As[0][(size_t)lr * LDSS + lc + 8 * q] = ra[q];
        *(bf16x8*)&Bs[0][(size_t)lr * LDSS + lc + 8 * q] = rb[q];
    }
#endif

    int cur = 0;
    for (int k0 = 0; k0 < K; k0 += BK) {
        const bool more = (k0 + BK) < K;

#if USE_TDM
        // issue next tile's DMA into the other buffer (freed by end barrier)
        if (more) {
            if (wave == 0)      tdm_load_tile(gAt + k0 + BK, lda, ldsA[cur ^ 1]);
            else if (wave == 1) tdm_load_tile(gBt + k0 + BK, ldb, ldsB[cur ^ 1]);
        }
        // tensor ops are in-order per wave: cnt<=1 means tile k0's DMA is done
        if (wave < 2) {
            if (more) __builtin_amdgcn_s_wait_tensorcnt(1);
            else      __builtin_amdgcn_s_wait_tensorcnt(0);
        }
        __syncthreads();   // buf[cur] full for all waves
#else
        if (more) {
#pragma unroll
            for (int q = 0; q < 4; q++) {
                ra[q] = *(const bf16x8*)(gA + k0 + BK + 8 * q);
                rb[q] = *(const bf16x8*)(gB + k0 + BK + 8 * q);
            }
            if (k0 + 2 * BK < K) {
                __builtin_prefetch(gA + k0 + 2 * BK, 0, 1);
                __builtin_prefetch(gB + k0 + 2 * BK, 0, 1);
            }
        }
        __syncthreads();   // buf[cur] ready for all waves
#endif

        // ---- compute on buf[cur]: 2 k-steps x (4x2) WMMAs -----------------
#pragma unroll
        for (int s = 0; s < 2; s++) {
            const int ks = s * 32;
            // A 16x32 frag: lanes<16 K=0..7,16..23 ; lanes>=16 K=8..15,24..31
            v16bf af[4];
#pragma unroll
            for (int i = 0; i < 4; i++) {
                const bf16* p = &As[cur][(size_t)(wm + 16 * i + lrow) * LDSS + ks + half * 8];
                bf16x8 lo = *(const bf16x8*)p;
                bf16x8 hi = *(const bf16x8*)(p + 16);
#pragma unroll
                for (int e = 0; e < 8; e++) { af[i][e] = lo[e]; af[i][8 + e] = hi[e]; }
            }
            // B 32x16 frag: col=lrow; lanes<16 K=0..7,16..23 ; lanes>=16 K=8..15,24..31
            v16bf bfv[2];
#pragma unroll
            for (int j = 0; j < 2; j++) {
                const bf16* p = &Bs[cur][(size_t)(wn + 16 * j + lrow) * LDSS + ks + half * 8];
                bf16x8 lo = *(const bf16x8*)p;
                bf16x8 hi = *(const bf16x8*)(p + 16);
#pragma unroll
                for (int e = 0; e < 8; e++) { bfv[j][e] = lo[e]; bfv[j][8 + e] = hi[e]; }
            }
#pragma unroll
            for (int i = 0; i < 4; i++)
#pragma unroll
                for (int j = 0; j < 2; j++)
                    acc[i][j] = __builtin_amdgcn_wmma_f32_16x16x32_bf16(
                        false, af[i], false, bfv[j], (short)0, acc[i][j], false, false);
        }

#if USE_TDM
        __syncthreads();   // all reads of buf[cur] done before next DMA hits it
#else
        if (more) {
#pragma unroll
            for (int q = 0; q < 4; q++) {
                *(bf16x8*)&As[cur ^ 1][(size_t)lr * LDSS + lc + 8 * q] = ra[q];
                *(bf16x8*)&Bs[cur ^ 1][(size_t)lr * LDSS + lc + 8 * q] = rb[q];
            }
        }
#endif
        cur ^= 1;
    }

    // ---- store: C 16x16 f32 layout: VGPR r -> M = r + 8*half, N = lrow ----
#pragma unroll
    for (int i = 0; i < 4; i++)
#pragma unroll
        for (int j = 0; j < 2; j++)
#pragma unroll
            for (int r = 0; r < 8; r++) {
                size_t row = blockM + wm + 16 * i + half * 8 + r;
                size_t col = blockN + wn + 16 * j + lrow;
                size_t idx = row * (size_t)ldc + col;
                if (accum) C[idx] += acc[i][j][r];
                else       C[idx]  = acc[i][j][r];
            }
}

// ---------------- elementwise / reduction kernels --------------------------

__global__ void cvt_bf16_kernel(const float* __restrict__ in, bf16* __restrict__ out, size_t n) {
    size_t i = (size_t)blockIdx.x * blockDim.x + threadIdx.x;
    if (i < n) out[i] = (bf16)in[i];
}

// W[K,N] f32 -> Wt[N,K] bf16 (one-time per call; output-coalesced)
__global__ void cvt_transpose_kernel(const float* __restrict__ in, bf16* __restrict__ out,
                                     int K, int N) {
    size_t i = (size_t)blockIdx.x * blockDim.x + threadIdx.x;
    size_t total = (size_t)K * N;
    if (i >= total) return;
    int n = (int)(i / K);
    int k = (int)(i % K);
    out[i] = (bf16)in[(size_t)k * N + n];
}

__global__ void lrelu_split_kernel(const float* __restrict__ x, bf16* __restrict__ phi,
                                   bf16* __restrict__ om, size_t n, int make_om) {
    size_t i = (size_t)blockIdx.x * blockDim.x + threadIdx.x;
    if (i >= n) return;
    float v = x[i];
    float f = v > 0.0f ? v : 0.01f * v;
    phi[i] = (bf16)f;
    if (make_om) om[i] = (bf16)(1.0f - f);
}

// per-column mean / rsqrt(var+eps) over M rows (biased variance, BN train mode)
__global__ void colstats_kernel(const float* __restrict__ X, int M, int ld,
                                float* __restrict__ mean, float* __restrict__ rstd) {
    int col = blockIdx.x;
    float s = 0.0f, ss = 0.0f;
    for (int r = threadIdx.x; r < M; r += blockDim.x) {
        float v = X[(size_t)r * ld + col];
        s += v; ss += v * v;
    }
    __shared__ float sh1[256], sh2[256];
    sh1[threadIdx.x] = s; sh2[threadIdx.x] = ss;
    __syncthreads();
    for (int st = 128; st > 0; st >>= 1) {
        if (threadIdx.x < st) { sh1[threadIdx.x] += sh1[threadIdx.x + st];
                                sh2[threadIdx.x] += sh2[threadIdx.x + st]; }
        __syncthreads();
    }
    if (threadIdx.x == 0) {
        float m = sh1[0] / M;
        float var = sh2[0] / M - m * m;
        mean[col] = m;
        rstd[col] = rsqrtf(var + 1e-5f);
    }
}

// F = tanh(m1*bn0(t0) + m2*bn2(t2))   [B,768]
__global__ void hgd_fuse_kernel(const float* __restrict__ m1, const float* __restrict__ m2,
                                const float* __restrict__ t0, const float* __restrict__ mu0,
                                const float* __restrict__ rs0, const float* __restrict__ g0,
                                const float* __restrict__ b0,
                                const float* __restrict__ t2, const float* __restrict__ mu2,
                                const float* __restrict__ rs2, const float* __restrict__ g2,
                                const float* __restrict__ b2,
                                float* __restrict__ F, size_t n, int cols) {
    size_t i = (size_t)blockIdx.x * blockDim.x + threadIdx.x;
    if (i >= n) return;
    int c = (int)(i % cols);
    float a  = g0[c] * (t0[i] - mu0[c]) * rs0[c] + b0[c];
    float bb = g2[c] * (t2[i] - mu2[c]) * rs2[c] + b2[c];
    F[i] = tanhf(m1[i] * a + m2[i] * bb);
}

// out[:, :D]=F_S ; out[:, D:2D]=F_C*(1-tpp). Writes f32 (optional) + bf16.
__global__ void hgd_tail_kernel(const float* __restrict__ FS, const float* __restrict__ FC,
                                const float* __restrict__ tpp,
                                float* __restrict__ outF, bf16* __restrict__ outBF,
                                size_t n, int D) {
    size_t i = (size_t)blockIdx.x * blockDim.x + threadIdx.x;
    if (i >= n) return;
    int cols = 2 * D;
    size_t row = i / cols;
    int    c   = (int)(i % cols);
    float v;
    if (c < D) v = FS[row * D + c];
    else { size_t k = row * D + (c - D); v = FC[k] * (1.0f - tpp[k]); }
    if (outF) outF[i] = v;
    outBF[i] = (bf16)v;
}

// rh = bf16( sigmoid(bn(x)) * H )
__global__ void bn_sigmoid_mul_cvt_kernel(const float* __restrict__ x, const float* __restrict__ mu,
                                          const float* __restrict__ rs, const float* __restrict__ g,
                                          const float* __restrict__ b, const float* __restrict__ H,
                                          bf16* __restrict__ rh, size_t n, int cols) {
    size_t i = (size_t)blockIdx.x * blockDim.x + threadIdx.x;
    if (i >= n) return;
    int c = (int)(i % cols);
    float t = g[c] * (x[i] - mu[c]) * rs[c] + b[c];
    float s = 1.0f / (1.0f + expf(-t));
    rh[i] = (bf16)(s * H[i]);
}

// Z = sigmoid(bn(Z)) in place
__global__ void bn_sigmoid_inplace_kernel(float* __restrict__ Z, const float* __restrict__ mu,
                                          const float* __restrict__ rs, const float* __restrict__ g,
                                          const float* __restrict__ b, size_t n, int cols) {
    size_t i = (size_t)blockIdx.x * blockDim.x + threadIdx.x;
    if (i >= n) return;
    int c = (int)(i % cols);
    float t = g[c] * (Z[i] - mu[c]) * rs[c] + b[c];
    Z[i] = 1.0f / (1.0f + expf(-t));
}

// H = Z*H + (1-Z)*tanh(hh) ; Hb = bf16(H)
__global__ void gru_update_kernel(float* __restrict__ H, const float* __restrict__ Z,
                                  const float* __restrict__ hh, bf16* __restrict__ Hb, size_t n) {
    size_t i = (size_t)blockIdx.x * blockDim.x + threadIdx.x;
    if (i >= n) return;
    float z = Z[i];
    float h = z * H[i] + (1.0f - z) * tanhf(hh[i]);
    H[i] = h;
    Hb[i] = (bf16)h;
}

// out[B,2] (+)= br[B,1536] @ w[1536,2]  (tiny GEMM, VALU is fine)
__global__ void out_acc_kernel(const float* __restrict__ br, const float* __restrict__ w,
                               float* __restrict__ out, int init, const float* __restrict__ bias) {
    int row = blockIdx.x * blockDim.x + threadIdx.x;
    if (row >= BBATCH) return;
    const float* r = br + (size_t)row * DD2;
    float a0 = 0.0f, a1 = 0.0f;
    for (int k = 0; k < DD2; k++) {
        float v = r[k];
        a0 += v * w[k * NLAB + 0];
        a1 += v * w[k * NLAB + 1];
    }
    if (init) { out[row * NLAB + 0] = bias[0] + a0; out[row * NLAB + 1] = bias[1] + a1; }
    else      { out[row * NLAB + 0] += a0;          out[row * NLAB + 1] += a1; }
}

// ---------------------------------------------------------------------------

extern "C" void kernel_launch(void* const* d_in, const int* in_sizes, int n_in,
                              void* d_out, int out_size, void* d_ws, size_t ws_size,
                              hipStream_t stream) {
    (void)in_sizes; (void)n_in; (void)out_size; (void)ws_size;

    // ---- input mapping (setup_inputs dict order) --------------------------
    const float* pooled = (const float*)d_in[0];
    int idx = 1;
    const float* Wsrc[2][9];
    const float* BNg[2][4];
    const float* BNb[2][4];
    for (int p = 0; p < 2; p++) {
        for (int k = 0; k < 9; k++) Wsrc[p][k] = (const float*)d_in[idx++];
        for (int k = 0; k < 4; k++) {
            BNg[p][k] = (const float*)d_in[idx++];
            BNb[p][k] = (const float*)d_in[idx++];
        }
    }
    const float* w_r = (const float*)d_in[idx++];
    const float* w_z = (const float*)d_in[idx++];
    const float* w_h = (const float*)d_in[idx++];
    const float* bn_r_g = (const float*)d_in[idx++];
    const float* bn_r_b = (const float*)d_in[idx++];
    const float* bn_z_g = (const float*)d_in[idx++];
    const float* bn_z_b = (const float*)d_in[idx++];
    const float* w_com[5];
    for (int k = 0; k < 5; k++) w_com[k] = (const float*)d_in[idx++];
    const float* out_w = (const float*)d_in[idx++];
    const float* out_b = (const float*)d_in[idx++];
    float* out = (float*)d_out;

    // ---- workspace carve (bump allocator) ---------------------------------
    char* base = (char*)d_ws;
    size_t off = 0;
    auto alloc = [&](size_t bytes) -> void* {
        off = (off + 255) & ~(size_t)255;
        void* p = base + off;
        off += bytes;
        return p;
    };

    auto blocks1 = [](size_t n) { return (unsigned)((n + 255) / 256); };

    auto cvt = [&](const float* src, bf16* dst, size_t n) {
        cvt_bf16_kernel<<<blocks1(n), 256, 0, stream>>>(src, dst, n);
    };
    auto cvtT = [&](const float* src, bf16* dst, int K, int N) {
        cvt_transpose_kernel<<<blocks1((size_t)K * N), 256, 0, stream>>>(src, dst, K, N);
    };
    // C[M,N] = A[M,K] x Wt[N,K] ; Wt row stride ldb (>= K for split weights)
    auto gemm = [&](const bf16* Ab, int lda, const bf16* Wt, int ldb,
                    float* C, int ldc, int M, int N, int K, int accum) {
        dim3 grid(M / BM, N / BN);
        gemm_bf16_kernel<<<grid, 256, 0, stream>>>(Ab, lda, Wt, ldb, C, ldc, K, accum);
    };

    // bf16 weights: converted + TRANSPOSED to [N,K] each call (cheap vs GEMMs)
    static const int wK[9] = {DD, DD, DD2, DD2, DD, DD, DD2, DD2, DD2};  // orig K
    bf16* wbf[2][9];
    for (int p = 0; p < 2; p++)
        for (int k = 0; k < 9; k++) {
            size_t sz = (size_t)wK[k] * DD;
            wbf[p][k] = (bf16*)alloc(sz * sizeof(bf16));
            cvtT(Wsrc[p][k], wbf[p][k], wK[k], DD);
        }
    const size_t WRZ = (size_t)DD4 * DD2;
    bf16* wr_bf = (bf16*)alloc(WRZ * sizeof(bf16)); cvtT(w_r, wr_bf, DD4, DD2);
    bf16* wz_bf = (bf16*)alloc(WRZ * sizeof(bf16)); cvtT(w_z, wz_bf, DD4, DD2);
    bf16* wh_bf = (bf16*)alloc(WRZ * sizeof(bf16)); cvtT(w_h, wh_bf, DD4, DD2);
    bf16* wcom_bf[5];
    for (int k = 0; k < 5; k++) { wcom_bf[k] = (bf16*)alloc(WRZ * sizeof(bf16)); cvtT(w_com[k], wcom_bf[k], DD4, DD2); }

    // activations / scratch
    bf16*  m1b     = (bf16*)alloc(BD * sizeof(bf16));
    bf16*  m2b     = (bf16*)alloc(BD * sizeof(bf16));
    float* phi_raw = (float*)alloc(BD2 * sizeof(float));
    bf16*  phi_bf  = (bf16*)alloc(BD2 * sizeof(bf16));
    bf16*  om_bf   = (bf16*)alloc(BD2 * sizeof(bf16));
    float* t0      = (float*)alloc(BD * sizeof(float));
    float* t2      = (float*)alloc(BD * sizeof(float));
    float* FC      = (float*)alloc(BD * sizeof(float));
    float* FS      = (float*)alloc(BD * sizeof(float));
    float* H       = (float*)alloc(BD2 * sizeof(float));
    bf16*  mp_bf   = (bf16*)alloc(BD2 * sizeof(bf16));
    bf16*  Xbf     = (bf16*)alloc(BD2 * sizeof(bf16));
    bf16*  Hbf     = (bf16*)alloc(BD2 * sizeof(bf16));
    float* raw     = (float*)alloc(BD2 * sizeof(float));
    float* Zt      = (float*)alloc(BD2 * sizeof(float));
    bf16*  rh_bf   = (bf16*)alloc(BD2 * sizeof(bf16));
    float* mu0 = (float*)alloc(DD2 * sizeof(float));
    float* rs0 = (float*)alloc(DD2 * sizeof(float));
    float* mu2 = (float*)alloc(DD2 * sizeof(float));
    float* rs2 = (float*)alloc(DD2 * sizeof(float));

    // ---- HGD block: m1,m2 [B,768] f32 -> out [B,1536] (f32 opt + bf16) ----
    auto run_hgd = [&](const float* m1, const float* m2, int p, float* outF, bf16* outBF) {
        cvt(m1, m1b, BD);
        cvt(m2, m2b, BD);
        // phi_c = lrelu([m1@gcb , m2@gca])
        gemm(m1b, DD, wbf[p][0], DD, phi_raw,       DD2, BBATCH, DD, DD, 0);
        gemm(m2b, DD, wbf[p][1], DD, phi_raw + DD,  DD2, BBATCH, DD, DD, 0);
        lrelu_split_kernel<<<blocks1(BD2), 256, 0, stream>>>(phi_raw, phi_bf, om_bf, BD2, 1);
        // t0 = phi@fcb ; t2 = (1-phi)@fca ; bn stats ; F_C
        gemm(phi_bf, DD2, wbf[p][2], DD2, t0, DD, BBATCH, DD, DD2, 0);
        gemm(om_bf,  DD2, wbf[p][3], DD2, t2, DD, BBATCH, DD, DD2, 0);
        colstats_kernel<<<DD, 256, 0, stream>>>(t0, BBATCH, DD, mu0, rs0);
        colstats_kernel<<<DD, 256, 0, stream>>>(t2, BBATCH, DD, mu2, rs2);
        hgd_fuse_kernel<<<blocks1(BD), 256, 0, stream>>>(
            m1, m2, t0, mu0, rs0, BNg[p][0], BNb[p][0],
                    t2, mu2, rs2, BNg[p][1], BNb[p][1], FC, BD, DD);
        // phi_s = lrelu([m1@gsb , m2@gsa])
        gemm(m1b, DD, wbf[p][4], DD, phi_raw,      DD2, BBATCH, DD, DD, 0);
        gemm(m2b, DD, wbf[p][5], DD, phi_raw + DD, DD2, BBATCH, DD, DD, 0);
        lrelu_split_kernel<<<blocks1(BD2), 256, 0, stream>>>(phi_raw, phi_bf, om_bf, BD2, 0);
        // s0 = phi_s@fsb ; s2 = phi_s@fsa ; F_S
        gemm(phi_bf, DD2, wbf[p][6], DD2, t0, DD, BBATCH, DD, DD2, 0);
        gemm(phi_bf, DD2, wbf[p][7], DD2, t2, DD, BBATCH, DD, DD2, 0);
        colstats_kernel<<<DD, 256, 0, stream>>>(t0, BBATCH, DD, mu0, rs0);
        colstats_kernel<<<DD, 256, 0, stream>>>(t2, BBATCH, DD, mu2, rs2);
        hgd_fuse_kernel<<<blocks1(BD), 256, 0, stream>>>(
            m1, m2, t0, mu0, rs0, BNg[p][2], BNb[p][2],
                    t2, mu2, rs2, BNg[p][3], BNb[p][3], FS, BD, DD);
        // tpp = phi_s@temp  (t0 reused)
        gemm(phi_bf, DD2, wbf[p][8], DD2, t0, DD, BBATCH, DD, DD2, 0);
        hgd_tail_kernel<<<blocks1(BD2), 256, 0, stream>>>(FS, FC, t0, outF, outBF, BD2, DD);
    };

    // ---- SGRU step: X=Xbf [B,1536], H (f32), HbIn (bf16 of H) -------------
    // split-concat GEMMs: Wt is [1536, 3072]; bottom half = column offset 1536
    auto sgru = [&](const bf16* Xb, const bf16* HbIn) {
        // Rt path
        gemm(Xb,   DD2, wr_bf,        DD4, raw, DD2, BBATCH, DD2, DD2, 0);
        gemm(HbIn, DD2, wr_bf + DD2,  DD4, raw, DD2, BBATCH, DD2, DD2, 1);
        colstats_kernel<<<DD2, 256, 0, stream>>>(raw, BBATCH, DD2, mu0, rs0);
        bn_sigmoid_mul_cvt_kernel<<<blocks1(BD2), 256, 0, stream>>>(
            raw, mu0, rs0, bn_r_g, bn_r_b, H, rh_bf, BD2, DD2);
        // Zt path
        gemm(Xb,   DD2, wz_bf,        DD4, Zt, DD2, BBATCH, DD2, DD2, 0);
        gemm(HbIn, DD2, wz_bf + DD2,  DD4, Zt, DD2, BBATCH, DD2, DD2, 1);
        colstats_kernel<<<DD2, 256, 0, stream>>>(Zt, BBATCH, DD2, mu2, rs2);
        bn_sigmoid_inplace_kernel<<<blocks1(BD2), 256, 0, stream>>>(
            Zt, mu2, rs2, bn_z_g, bn_z_b, BD2, DD2);
        // Hh = tanh([X, Rt*H] @ w_h)
        gemm(Xb,    DD2, wh_bf,       DD4, raw, DD2, BBATCH, DD2, DD2, 0);
        gemm(rh_bf, DD2, wh_bf + DD2, DD4, raw, DD2, BBATCH, DD2, DD2, 1);
        gru_update_kernel<<<blocks1(BD2), 256, 0, stream>>>(H, Zt, raw, Hbf, BD2);
    };

    // ---- full model -------------------------------------------------------
    const float* post = pooled;
    for (int n = 0; n < 5; n++) {
        const float* com = pooled + (size_t)(1 + 4 * n) * BD;
        // m_p = HGD_post(post, com); h init = m_p
        run_hgd(post, com, 0, H, mp_bf);
        for (int j = 0; j < 3; j++) {
            const float* sub = pooled + (size_t)(2 + 4 * n + j) * BD;
            run_hgd(com, sub, 1, nullptr, Xbf);          // m_c (bf16)
            sgru(Xbf, (j == 0) ? mp_bf : Hbf);           // h = SGRU(m_c, h)
        }
        // branch = [m_p, h] @ w_com[n]
        gemm(mp_bf, DD2, wcom_bf[n],       DD4, raw, DD2, BBATCH, DD2, DD2, 0);
        gemm(Hbf,   DD2, wcom_bf[n] + DD2, DD4, raw, DD2, BBATCH, DD2, DD2, 1);
        // out (+)= branch @ out_w[1536n:1536(n+1), :]  (+ bias on first)
        out_acc_kernel<<<blocks1(BBATCH), 256, 0, stream>>>(
            raw, out_w + (size_t)n * DD2 * NLAB, out, (n == 0) ? 1 : 0, out_b);
    }
}
